// Decoder_1_29798483099790
// MI455X (gfx1250) — matrix operand order repaired and checked by reading, not compiled, not run on hardware
//
#include <hip/hip_runtime.h>

// ---- CDNA5 / gfx1250: wave32, WMMA 16x16x32 f16 -> f32 ----
typedef __attribute__((ext_vector_type(16))) _Float16 v16h;
typedef __attribute__((ext_vector_type(8)))  float    v8f;
typedef __attribute__((ext_vector_type(4)))  float    v4f;

#define BATCH 32
#define LAT   64
#define HDIM  96
#define G4    384          // 4*H
#define TSEQ  512
#define VCAB  10000
#define KC_H  3            // K chunks of 32 over HDIM=96
#define NT_U  24           // 384/16 N-tiles for U
#define NT_V  625          // 10000/16 N-tiles for Wd
#define MT_A  1024         // 16384/16 M-tiles for hs

__device__ __forceinline__ v8f wmma16x16x32(v16h a, v16h b, v8f c) {
  // emits v_wmma_f32_16x16x32_f16
  return __builtin_amdgcn_wmma_f32_16x16x32_f16(false, a, false, b,
                                                (short)0, c, false, false);
}
// v_rcp_f32 (TRANS) instead of IEEE divide: keeps the sequential LSTM
// elementwise phase off the v_div_scale/v_div_fmas path.
__device__ __forceinline__ float rcp_f(float x) { return __builtin_amdgcn_rcpf(x); }
__device__ __forceinline__ float sigm(float x) { return rcp_f(1.0f + __expf(-x)); }
__device__ __forceinline__ float tanh_f(float x) {
  float e = __expf(2.0f * x);          // +inf / 0 at extremes -> exact +-1
  return 1.0f - 2.0f * rcp_f(e + 1.0f);
}

// ---------------- zp = z @ W + b  (tiny) ----------------
__global__ void k_zp(const float* __restrict__ z, const float* __restrict__ W,
                     const float* __restrict__ b, float* __restrict__ zp) {
  int idx = blockIdx.x * blockDim.x + threadIdx.x;
  if (idx >= BATCH * G4) return;
  int bb = idx / G4, n = idx % G4;
  float acc = b[n];
  for (int k = 0; k < LAT; ++k) acc = fmaf(z[bb * LAT + k], W[k * G4 + n], acc);
  zp[idx] = acc;
}

// ---- split f32 matrix [HDIM x ncols] into hi/lo f16 B-fragments ----
// B 32x16 f16 layout: N = lane&15; lane 0-15: K={0..7,16..23}; lane 16-31: K={8..15,24..31}.
__global__ void k_pack_b(const float* __restrict__ src, int ncols, int ntiles,
                         _Float16* __restrict__ dhi, _Float16* __restrict__ dlo) {
  int idx = blockIdx.x * blockDim.x + threadIdx.x;
  if (idx >= ntiles * KC_H * 32 * 16) return;
  int q = idx & 15, L = (idx >> 4) & 31, f = idx >> 9;
  int kc = f % KC_H, nt = f / KC_H;
  int n = nt * 16 + (L & 15);
  int k = kc * 32 + ((q < 8) ? (((L >= 16) ? 8 : 0) + q) : (16 + ((L >= 16) ? 8 : 0) + (q - 8)));
  float x = src[k * ncols + n];
  _Float16 h = (_Float16)x;
  dhi[idx] = h;
  dlo[idx] = (_Float16)(x - (float)h);
}

// ---- split hs [16384 x 96] into hi/lo f16 A-fragments ----
// A 16x32 f16 layout: M = L&15; L<16: K'={0..7,16..23}; L>=16: K'={8..15,24..31}.
__global__ void k_pack_a(const float* __restrict__ hs,
                         _Float16* __restrict__ dhi, _Float16* __restrict__ dlo) {
  int idx = blockIdx.x * blockDim.x + threadIdx.x;
  if (idx >= MT_A * KC_H * 32 * 16) return;
  int q = idx & 15, L = (idx >> 4) & 31, f = idx >> 9;
  int kc = f % KC_H, m = f / KC_H;
  int row = m * 16 + (L & 15);
  int hi16 = (L >= 16) ? 8 : 0;
  int k = kc * 32 + ((q < 8) ? (hi16 + q) : (16 + hi16 + (q - 8)));
  float x = hs[row * HDIM + k];
  _Float16 h = (_Float16)x;
  dhi[idx] = h;
  dlo[idx] = (_Float16)(x - (float)h);
}

// ---------------- LSTM: 2 blocks x 384 threads, 512 sequential steps ----------------
__global__ void __launch_bounds__(384, 1)
k_lstm(const float* __restrict__ zp, const _Float16* __restrict__ uhi,
       const _Float16* __restrict__ ulo, float* __restrict__ hs) {
  extern __shared__ char smem[];
  float*    gbuf = (float*)smem;                      // 16*384 f32 = 24576 B
  _Float16* ahi  = (_Float16*)(smem + 16 * G4 * 4);   // 3*512 halves = 3072 B
  _Float16* alo  = ahi + KC_H * 512;                  // 3072 B

  const int tid = threadIdx.x, lane = tid & 31, wave = tid >> 5;
  const int b0 = blockIdx.x * 16;                     // batch rows b0..b0+15
  const int rowoff = (lane >= 16) ? 8 : 0;            // C-layout M offset
  const int ncol = lane & 15;

  // U fragments for this wave's two N-tiles stay in VGPRs for all 512 steps.
  v16h ub[2][KC_H][2];
  v8f  zpf[2];
#pragma unroll
  for (int tt = 0; tt < 2; ++tt) {
    int nt = wave * 2 + tt;
#pragma unroll
    for (int kc = 0; kc < KC_H; ++kc) {
      int base = ((nt * KC_H + kc) * 32 + lane) * 16;
      ub[tt][kc][0] = *(const v16h*)(uhi + base);
      ub[tt][kc][1] = *(const v16h*)(ulo + base);
    }
#pragma unroll
    for (int r = 0; r < 8; ++r)
      zpf[tt][r] = zp[(b0 + r + rowoff) * G4 + nt * 16 + ncol];
  }

  // Elementwise ownership: 4 consecutive h columns of one batch row; c in regs.
  const int erow = tid / 24;
  const int ecol = (tid % 24) * 4;
  float cst[4] = {0.f, 0.f, 0.f, 0.f};

  for (int i = tid; i < KC_H * 512; i += 384) { ahi[i] = (_Float16)0.f; alo[i] = (_Float16)0.f; }
  __syncthreads();

  for (int t = 0; t < TSEQ; ++t) {
    // --- gate GEMM: g = zp + h @ U ---
    v16h af[KC_H][2];
#pragma unroll
    for (int kc = 0; kc < KC_H; ++kc) {
      af[kc][0] = *(const v16h*)(ahi + kc * 512 + lane * 16);
      af[kc][1] = *(const v16h*)(alo + kc * 512 + lane * 16);
    }
    v8f acc[2] = {zpf[0], zpf[1]};
#pragma unroll
    for (int kc = 0; kc < KC_H; ++kc) {
#pragma unroll
      for (int tt = 0; tt < 2; ++tt) {
        acc[tt] = wmma16x16x32(af[kc][0], ub[tt][kc][0], acc[tt]);
        acc[tt] = wmma16x16x32(af[kc][0], ub[tt][kc][1], acc[tt]);
        acc[tt] = wmma16x16x32(af[kc][1], ub[tt][kc][0], acc[tt]);
      }
    }
#pragma unroll
    for (int tt = 0; tt < 2; ++tt) {
      int nt = wave * 2 + tt;
#pragma unroll
      for (int r = 0; r < 8; ++r)
        gbuf[(r + rowoff) * G4 + nt * 16 + ncol] = acc[tt][r];
    }
    __syncthreads();

    // --- elementwise gates, c/h update, re-split h into A fragments ---
    v4f hv;
#pragma unroll
    for (int j = 0; j < 4; ++j) {
      int col = ecol + j;
      float gi = gbuf[erow * G4 + col];
      float gf = gbuf[erow * G4 + 96 + col];
      float gc = gbuf[erow * G4 + 192 + col];
      float go = gbuf[erow * G4 + 288 + col];
      float c = sigm(gf) * cst[j] + sigm(gi) * tanh_f(gc);
      float h = sigm(go) * tanh_f(c);
      cst[j] = c;
      hv[j] = h;
      int kc = col >> 5, kk = col & 31;
      int s = (kk >> 3) & 1;
      int q = (kk & 7) + ((kk & 16) ? 8 : 0);
      int fi = kc * 512 + (erow + 16 * s) * 16 + q;
      _Float16 hh = (_Float16)h;
      ahi[fi] = hh;
      alo[fi] = (_Float16)(h - (float)hh);
    }
    *(v4f*)(hs + ((size_t)(b0 + erow) * TSEQ + t) * HDIM + ecol) = hv;
    __syncthreads();
  }
}

// ---------------- logits = hs @ Wd + bd  (bulk: 31.5 GFLOP) ----------------
__global__ void __launch_bounds__(256, 1)
k_gemm(const _Float16* __restrict__ ahi, const _Float16* __restrict__ alo,
       const _Float16* __restrict__ bhi, const _Float16* __restrict__ blo,
       const float* __restrict__ bd, float* __restrict__ out) {
  const int lane = threadIdx.x & 31, wave = threadIdx.x >> 5;
  const int m = blockIdx.x;
  const int ncol = lane & 15;
  const int rowbase = m * 16 + ((lane >= 16) ? 8 : 0);

  v16h A[KC_H][2];
#pragma unroll
  for (int kc = 0; kc < KC_H; ++kc) {
    int base = ((m * KC_H + kc) * 32 + lane) * 16;
    A[kc][0] = *(const v16h*)(ahi + base);
    A[kc][1] = *(const v16h*)(alo + base);
  }

  for (int nt0 = wave; nt0 < NT_V; nt0 += 16) {
    const int nt1 = nt0 + 8;
    const bool has1 = (nt1 < NT_V);
    if (nt0 + 16 < NT_V)  // warm L2/L0 for next trip
      __builtin_prefetch(bhi + (size_t)(nt0 + 16) * KC_H * 512 + lane * 16, 0, 1);

    v16h B0[KC_H][2], B1[KC_H][2];
#pragma unroll
    for (int kc = 0; kc < KC_H; ++kc) {
      int base0 = ((nt0 * KC_H + kc) * 32 + lane) * 16;
      B0[kc][0] = *(const v16h*)(bhi + base0);
      B0[kc][1] = *(const v16h*)(blo + base0);
    }
    if (has1) {
#pragma unroll
      for (int kc = 0; kc < KC_H; ++kc) {
        int base1 = ((nt1 * KC_H + kc) * 32 + lane) * 16;
        B1[kc][0] = *(const v16h*)(bhi + base1);
        B1[kc][1] = *(const v16h*)(blo + base1);
      }
    }
    v8f acc0 = {}, acc1 = {};
#pragma unroll
    for (int kc = 0; kc < KC_H; ++kc) {
      acc0 = wmma16x16x32(A[kc][0], B0[kc][0], acc0);
      if (has1) acc1 = wmma16x16x32(A[kc][0], B1[kc][0], acc1);
      acc0 = wmma16x16x32(A[kc][0], B0[kc][1], acc0);
      if (has1) acc1 = wmma16x16x32(A[kc][0], B1[kc][1], acc1);
      acc0 = wmma16x16x32(A[kc][1], B0[kc][0], acc0);
      if (has1) acc1 = wmma16x16x32(A[kc][1], B1[kc][0], acc1);
    }
    float bias0 = bd[nt0 * 16 + ncol];
#pragma unroll
    for (int r = 0; r < 8; ++r)
      out[(size_t)(rowbase + r) * VCAB + nt0 * 16 + ncol] = acc0[r] + bias0;
    if (has1) {
      float bias1 = bd[nt1 * 16 + ncol];
#pragma unroll
      for (int r = 0; r < 8; ++r)
        out[(size_t)(rowbase + r) * VCAB + nt1 * 16 + ncol] = acc1[r] + bias1;
    }
  }
}

// ---------------- row softmax, register-resident (1 read + 1 write) ----------------
__global__ void __launch_bounds__(256, 1) k_softmax(float* __restrict__ out) {
  __shared__ float red[256];
  const int tid = threadIdx.x;
  float* p = out + (size_t)blockIdx.x * VCAB;

  float v[40];
  float mx = -3.4e38f;
#pragma unroll
  for (int j = 0; j < 40; ++j) {
    int i = tid + j * 256;
    v[j] = (i < VCAB) ? p[i] : -3.4e38f;
    mx = fmaxf(mx, v[j]);
  }
  red[tid] = mx; __syncthreads();
  for (int s = 128; s > 0; s >>= 1) {
    if (tid < s) red[tid] = fmaxf(red[tid], red[tid + s]);
    __syncthreads();
  }
  mx = red[0]; __syncthreads();

  float sum = 0.f;
#pragma unroll
  for (int j = 0; j < 40; ++j) { v[j] = __expf(v[j] - mx); sum += v[j]; }
  red[tid] = sum; __syncthreads();
  for (int s = 128; s > 0; s >>= 1) {
    if (tid < s) red[tid] += red[tid + s];
    __syncthreads();
  }
  float inv = rcp_f(red[0]);
#pragma unroll
  for (int j = 0; j < 40; ++j) {
    int i = tid + j * 256;
    if (i < VCAB) p[i] = v[j] * inv;
  }
}

extern "C" void kernel_launch(void* const* d_in, const int* in_sizes, int n_in,
                              void* d_out, int out_size, void* d_ws, size_t ws_size,
                              hipStream_t stream) {
  (void)in_sizes; (void)n_in; (void)out_size; (void)ws_size;
  const float* z  = (const float*)d_in[0];
  const float* W  = (const float*)d_in[1];
  const float* U  = (const float*)d_in[2];
  const float* b  = (const float*)d_in[3];
  const float* Wd = (const float*)d_in[4];
  const float* bd = (const float*)d_in[5];
  float* out = (float*)d_out;

  // workspace carve (all offsets 32B-aligned)
  char* w = (char*)d_ws;
  float*    zp  = (float*)w;     w += (size_t)BATCH * G4 * 4;            // 48 KB
  float*    hs  = (float*)w;     w += (size_t)BATCH * TSEQ * HDIM * 4;   // 6.29 MB
  _Float16* uhi = (_Float16*)w;  w += (size_t)NT_U * KC_H * 512 * 2;     // 72 KB
  _Float16* ulo = (_Float16*)w;  w += (size_t)NT_U * KC_H * 512 * 2;
  _Float16* bhi = (_Float16*)w;  w += (size_t)NT_V * KC_H * 512 * 2;     // 1.92 MB
  _Float16* blo = (_Float16*)w;  w += (size_t)NT_V * KC_H * 512 * 2;
  _Float16* afh = (_Float16*)w;  w += (size_t)MT_A * KC_H * 512 * 2;     // 3.15 MB
  _Float16* afl = (_Float16*)w;  w += (size_t)MT_A * KC_H * 512 * 2;

  k_zp<<<(BATCH * G4 + 255) / 256, 256, 0, stream>>>(z, W, b, zp);
  k_pack_b<<<(NT_U * KC_H * 512 + 255) / 256, 256, 0, stream>>>(U, G4, NT_U, uhi, ulo);
  k_pack_b<<<(NT_V * KC_H * 512 + 255) / 256, 256, 0, stream>>>(Wd, VCAB, NT_V, bhi, blo);
  k_lstm<<<2, 384, 16 * G4 * 4 + 2 * KC_H * 512 * 2, stream>>>(zp, uhi, ulo, hs);
  k_pack_a<<<(MT_A * KC_H * 512 + 255) / 256, 256, 0, stream>>>(hs, afh, afl);
  k_gemm<<<MT_A, 256, 0, stream>>>(afh, afl, bhi, blo, bd, out);
  k_softmax<<<BATCH * TSEQ, 256, 0, stream>>>(out);
}
